// CustomMHA_30717606101336
// MI455X (gfx1250) — matrix operand order repaired and checked by reading, not compiled, not run on hardware
//
#include <hip/hip_runtime.h>
#include <hip/hip_bf16.h>

typedef __bf16 bf16_t;
typedef __attribute__((ext_vector_type(16))) __bf16 v16bf;
typedef __attribute__((ext_vector_type(8)))  __bf16 v8bf;
typedef __attribute__((ext_vector_type(8)))  float  v8f;

constexpr int  Bb = 4, Tt = 2048, Dd = 1024, Hh = 16, Dh = 64;
constexpr long MT = (long)Bb * Tt;           // 8192 tokens
constexpr long XN = MT * Dd;                 // 8388608
constexpr long WN = (long)Dd * Dd;           // 1048576

// ---------------------------------------------------------------- fragments
// 16-bit A 16x32 layout (ISA 7.12.2): lanes 0-15 row M=lm hold K k0+{0..7}
// in v[0..3] and k0+16+{0..7} in v[4..7]; lanes 16-31 hold K +8.
__device__ inline v16bf load_a_frag_bf(const bf16_t* base, long ld, int row0,
                                       int k0, int lane) {
  int hl = lane >> 4, lm = lane & 15;
  const bf16_t* p = base + (long)(row0 + lm) * ld + k0 + hl * 8;
  v8bf lo = *(const v8bf*)p;
  v8bf hi = *(const v8bf*)(p + 16);
  v16bf r;
#pragma unroll
  for (int i = 0; i < 8; i++) { r[i] = lo[i]; r[8 + i] = hi[i]; }
  return r;
}

__device__ inline v16bf load_a_frag_f32(const float* base, long ld, int row0,
                                        int k0, int lane) {
  int hl = lane >> 4, lm = lane & 15;
  const float* p = base + (long)(row0 + lm) * ld + k0 + hl * 8;
  v8f lo = *(const v8f*)p;
  v8f hi = *(const v8f*)(p + 16);
  v16bf r;
#pragma unroll
  for (int i = 0; i < 8; i++) { r[i] = (bf16_t)lo[i]; r[8 + i] = (bf16_t)hi[i]; }
  return r;
}

// 16-bit B 32x16 layout: lane n (0-15) = column col0+n, K k0+{0..7} in
// v[0..3] and k0+16+{0..7} in v[4..7]; lanes 16-31 hold K +8.  Two 16B loads.
__device__ inline v16bf load_b_frag_bf(const bf16_t* base, long ld, int col0,
                                       int k0, int lane) {
  int hl = lane >> 4, lm = lane & 15;
  const bf16_t* p = base + (long)(col0 + lm) * ld + k0 + hl * 8;
  v8bf lo = *(const v8bf*)p;
  v8bf hi = *(const v8bf*)(p + 16);
  v16bf r;
#pragma unroll
  for (int i = 0; i < 8; i++) { r[i] = lo[i]; r[8 + i] = hi[i]; }
  return r;
}

__device__ inline v8f wmma_bf16(v16bf a, v16bf b, v8f c) {
  return __builtin_amdgcn_wmma_f32_16x16x32_bf16(false, a, false, b,
                                                 (short)0, c, false, false);
}

// ---------------------------------------------------------------- convert
__global__ void cvt_f32_bf16_kernel(const float* __restrict__ s,
                                    bf16_t* __restrict__ d, long n) {
  long i = (long)blockIdx.x * blockDim.x + threadIdx.x;
  long stride = (long)gridDim.x * blockDim.x;
  for (; i < n; i += stride) d[i] = (bf16_t)s[i];
}

// ---------------------------------------------------------------- generic GEMM
// C[M,N] = Arows[M,K] . Brows[N,K]^T (+bias).  BIAS_MODE: 0 none, 1 per-col,
// 2 per-row.  Each wave owns a 64x64 tile (4x4 accumulators).
template <typename OutT, int BIAS_MODE>
__global__ __launch_bounds__(256) void gemm_bf16_kernel(
    const bf16_t* __restrict__ A, long lda, const bf16_t* __restrict__ Bm,
    long ldb, const float* __restrict__ bias, OutT* __restrict__ C, long ldc,
    int M, int N, int Kd) {
  int lane = threadIdx.x & 31;
  int wave = threadIdx.x >> 5;
  int ntn = N >> 6;
  long tile = (long)blockIdx.x * (blockDim.x >> 5) + wave;
  long ntiles = (long)(M >> 6) * ntn;
  if (tile >= ntiles) return;                     // wave-uniform exit
  int tm = (int)(tile / ntn) * 64;
  int tn = (int)(tile % ntn) * 64;

  v8f acc[4][4] = {};
  for (int k0 = 0; k0 < Kd; k0 += 32) {
    v16bf af[4], bfm[4];
#pragma unroll
    for (int i = 0; i < 4; i++) af[i] = load_a_frag_bf(A, lda, tm + 16 * i, k0, lane);
#pragma unroll
    for (int j = 0; j < 4; j++) bfm[j] = load_b_frag_bf(Bm, ldb, tn + 16 * j, k0, lane);
#pragma unroll
    for (int i = 0; i < 4; i++)
#pragma unroll
      for (int j = 0; j < 4; j++) acc[i][j] = wmma_bf16(af[i], bfm[j], acc[i][j]);
  }

  int hl = lane >> 4, lm = lane & 15;
#pragma unroll
  for (int i = 0; i < 4; i++)
#pragma unroll
    for (int j = 0; j < 4; j++) {
      int col = tn + 16 * j + lm;
      float bc = (BIAS_MODE == 1) ? bias[col] : 0.f;
#pragma unroll
      for (int r = 0; r < 8; r++) {
        int row = tm + 16 * i + hl * 8 + r;
        float v = acc[i][j][r] + bc;
        if (BIAS_MODE == 2) v += bias[row];
        C[(long)row * ldc + col] = (OutT)v;
      }
    }
}

// ---------------------------------------------------------------- scores
// S[b,h,t,s] = (q . k) * 0.125 ; per (b,h) M=N=T, K=Dh=64, fp32 out.
__global__ __launch_bounds__(256) void attn_scores_kernel(
    const bf16_t* __restrict__ Q, const bf16_t* __restrict__ K,
    float* __restrict__ S) {
  int lane = threadIdx.x & 31;
  int wave = threadIdx.x >> 5;
  int bh = blockIdx.y, b = bh >> 4, h = bh & 15;
  int tile = blockIdx.x * 8 + wave;               // 0..1023
  const int TPN = Tt / 64;                        // 32
  int tm = (tile / TPN) * 64, tn = (tile % TPN) * 64;
  const bf16_t* qb = Q + (long)b * Tt * Dd + h * Dh;
  const bf16_t* kb = K + (long)b * Tt * Dd + h * Dh;
  float* outp = S + (long)bh * Tt * Tt;

  v8f acc[4][4] = {};
#pragma unroll
  for (int k0 = 0; k0 < Dh; k0 += 32) {
    v16bf af[4], bfm[4];
#pragma unroll
    for (int i = 0; i < 4; i++) af[i] = load_a_frag_bf(qb, Dd, tm + 16 * i, k0, lane);
#pragma unroll
    for (int j = 0; j < 4; j++) bfm[j] = load_b_frag_bf(kb, Dd, tn + 16 * j, k0, lane);
#pragma unroll
    for (int i = 0; i < 4; i++)
#pragma unroll
      for (int j = 0; j < 4; j++) acc[i][j] = wmma_bf16(af[i], bfm[j], acc[i][j]);
  }
  int hl = lane >> 4, lm = lane & 15;
#pragma unroll
  for (int i = 0; i < 4; i++)
#pragma unroll
    for (int j = 0; j < 4; j++) {
      int col = tn + 16 * j + lm;
#pragma unroll
      for (int r = 0; r < 8; r++) {
        int row = tm + 16 * i + hl * 8 + r;
        outp[(long)row * Tt + col] = acc[i][j][r] * 0.125f;
      }
    }
}

// ---------------------------------------------------------------- softmax+gate
__global__ __launch_bounds__(256) void softmax_gate_kernel(
    float* __restrict__ A, const float* __restrict__ gates) {
  __shared__ float red[256];
  int tid = threadIdx.x;
  long rid = blockIdx.x;                          // 0 .. B*H*T-1
  int bh = (int)(rid / Tt);
  float* row = A + rid * Tt;
  float v[8];
  float m = -3.4e38f;
#pragma unroll
  for (int i = 0; i < 8; i++) { v[i] = row[tid + 256 * i]; m = fmaxf(m, v[i]); }
  red[tid] = m; __syncthreads();
  for (int s = 128; s > 0; s >>= 1) {
    if (tid < s) red[tid] = fmaxf(red[tid], red[tid + s]);
    __syncthreads();
  }
  float rm = red[0]; __syncthreads();
  float sum = 0.f;
#pragma unroll
  for (int i = 0; i < 8; i++) { v[i] = __expf(v[i] - rm); sum += v[i]; }
  red[tid] = sum; __syncthreads();
  for (int s = 128; s > 0; s >>= 1) {
    if (tid < s) red[tid] += red[tid + s];
    __syncthreads();
  }
  float g = gates[bh] / red[0];
#pragma unroll
  for (int i = 0; i < 8; i++) row[tid + 256 * i] = v[i] * g;
}

// ---------------------------------------------------------------- z = A . V
// Per (b,h): M=T, N=Dh, K=T.  A fp32 (converted to bf16 in regs), Vt is
// [D, B*T] (row d contiguous in token s) so B-frags are contiguous loads.
__global__ __launch_bounds__(256) void attn_z_kernel(
    const float* __restrict__ A, const bf16_t* __restrict__ Vt,
    bf16_t* __restrict__ Z) {
  int lane = threadIdx.x & 31;
  int wave = threadIdx.x >> 5;
  int bh = blockIdx.y, b = bh >> 4, h = bh & 15;
  int tile = blockIdx.x * 8 + wave;               // 0..31
  int tm = tile * 64;
  const float* Ab = A + (long)bh * Tt * Tt;
  const bf16_t* Vb = Vt + (long)h * Dh * MT + (long)b * Tt;

  v8f acc[4][4] = {};
  for (int k0 = 0; k0 < Tt; k0 += 32) {
    v16bf af[4], bfm[4];
#pragma unroll
    for (int i = 0; i < 4; i++) af[i] = load_a_frag_f32(Ab, Tt, tm + 16 * i, k0, lane);
#pragma unroll
    for (int j = 0; j < 4; j++) bfm[j] = load_b_frag_bf(Vb, MT, 16 * j, k0, lane);
#pragma unroll
    for (int i = 0; i < 4; i++)
#pragma unroll
      for (int j = 0; j < 4; j++) acc[i][j] = wmma_bf16(af[i], bfm[j], acc[i][j]);
  }
  int hl = lane >> 4, lm = lane & 15;
#pragma unroll
  for (int i = 0; i < 4; i++)
#pragma unroll
    for (int j = 0; j < 4; j++) {
      int col = 16 * j + lm;                      // 0..63 within head
#pragma unroll
      for (int r = 0; r < 8; r++) {
        int row = tm + 16 * i + hl * 8 + r;       // token t
        Z[((long)b * Tt + row) * Dd + h * Dh + col] = (bf16_t)acc[i][j][r];
      }
    }
}

// ---------------------------------------------------------------- host
extern "C" void kernel_launch(void* const* d_in, const int* in_sizes, int n_in,
                              void* d_out, int out_size, void* d_ws,
                              size_t ws_size, hipStream_t stream) {
  const float* x  = (const float*)d_in[0];
  const float* gates = (const float*)d_in[1];
  const float* Wq = (const float*)d_in[2]; const float* bq = (const float*)d_in[3];
  const float* Wk = (const float*)d_in[4]; const float* bk = (const float*)d_in[5];
  const float* Wv = (const float*)d_in[6]; const float* bv = (const float*)d_in[7];
  const float* Wo = (const float*)d_in[8]; const float* bo = (const float*)d_in[9];

  float* out  = (float*)d_out;                    // [B,T,D]
  float* Aout = out + XN;                         // [B,H,T,T]

  bf16_t* ws  = (bf16_t*)d_ws;                    // ~88 MB of bf16 scratch
  bf16_t* xbf = ws;
  bf16_t* wqb = xbf + XN;
  bf16_t* wkb = wqb + WN;
  bf16_t* wvb = wkb + WN;
  bf16_t* wob = wvb + WN;
  bf16_t* Qb  = wob + WN;                         // [M, D]
  bf16_t* Kb  = Qb + XN;                          // [M, D]
  bf16_t* Vt  = Kb + XN;                          // [D, M]  (transposed V)
  bf16_t* Zb  = Vt + XN;                          // [M, D]

  cvt_f32_bf16_kernel<<<4096, 256, 0, stream>>>(x,  xbf, XN);
  cvt_f32_bf16_kernel<<<1024, 256, 0, stream>>>(Wq, wqb, WN);
  cvt_f32_bf16_kernel<<<1024, 256, 0, stream>>>(Wk, wkb, WN);
  cvt_f32_bf16_kernel<<<1024, 256, 0, stream>>>(Wv, wvb, WN);
  cvt_f32_bf16_kernel<<<1024, 256, 0, stream>>>(Wo, wob, WN);

  // Q = X.Wq^T + bq ; K = X.Wk^T + bk     (2048 tiles -> 256 blocks x 8 waves)
  gemm_bf16_kernel<bf16_t, 1><<<256, 256, 0, stream>>>(
      xbf, Dd, wqb, Dd, bq, Qb, Dd, (int)MT, Dd, Dd);
  gemm_bf16_kernel<bf16_t, 1><<<256, 256, 0, stream>>>(
      xbf, Dd, wkb, Dd, bk, Kb, Dd, (int)MT, Dd, Dd);
  // Vt = Wv.X^T + bv (bias on rows): output [D, M] so A.V has contiguous B.
  gemm_bf16_kernel<bf16_t, 2><<<256, 256, 0, stream>>>(
      wvb, Dd, xbf, Dd, bv, Vt, MT, Dd, (int)MT, Dd);

  // raw scaled scores into d_out's A region
  attn_scores_kernel<<<dim3(128, Bb * Hh), 256, 0, stream>>>(Qb, Kb, Aout);
  // softmax + gate in place -> final A
  softmax_gate_kernel<<<(int)(MT * Hh), 256, 0, stream>>>(Aout, gates);
  // z = A . V
  attn_z_kernel<<<dim3(4, Bb * Hh), 256, 0, stream>>>(Aout, Vt, Zb);
  // out = z.Wo^T + bo  (fp32)
  gemm_bf16_kernel<float, 1><<<256, 256, 0, stream>>>(
      Zb, Dd, wob, Dd, bo, out, Dd, (int)MT, Dd, Dd);
}
